// VisionMamba_17119739642491
// MI455X (gfx1250) — compile-verified, matrix operand order checked by reading, not run on hardware
//
#include <hip/hip_runtime.h>

#define LSEQ   1569
#define NB     2
#define BL     (NB * LSEQ)      // 3138 valid rows (b*LSEQ + l)
#define MPAD   3200             // 200 tiles of 16, divisible by 4 for M-blocking
#define DM     192
#define DI     384
#define DXZ    768
#define NXP    48               // xproj out dim padded 44 -> 48
#define DTR    12
#define DST    16
#define DEPTH  12

typedef __attribute__((ext_vector_type(16))) _Float16 v16h;
typedef __attribute__((ext_vector_type(8)))  _Float16 v8h;
typedef __attribute__((ext_vector_type(8)))  float    v8f;

// ---------------------------------------------------------------- utilities
__global__ void k_cvt_f16(const float* __restrict__ src, _Float16* __restrict__ dst, int n) {
  int i = blockIdx.x * blockDim.x + threadIdx.x;
  if (i < n) dst[i] = (_Float16)src[i];
}

// xproj (DEPTH,44,384) -> f16 (DEPTH,48,384), pad rows zero
__global__ void k_cvt_xproj(const float* __restrict__ src, _Float16* __restrict__ dst) {
  int i = blockIdx.x * blockDim.x + threadIdx.x;
  if (i >= DEPTH * NXP * DI) return;
  int c = i % DI, r = (i / DI) % NXP, lay = i / (DI * NXP);
  dst[i] = (r < 44) ? (_Float16)src[((size_t)lay * 44 + r) * DI + c] : (_Float16)0.f;
}

// ------------------------------------------------------------- patch embed
__global__ void k_patch_embed(const float* __restrict__ x, const float* __restrict__ pw,
                              const float* __restrict__ pb, const float* __restrict__ cls,
                              const float* __restrict__ pos, const float* __restrict__ tpos,
                              float* __restrict__ h) {
  int i = blockIdx.x * blockDim.x + threadIdx.x;
  if (i >= BL * DM) return;
  int d = i % DM, row = i / DM, b = row / LSEQ, l = row % LSEQ;
  float v;
  if (l == 0) {
    v = cls[d] + pos[d];
  } else {
    int p = l - 1, t = p / 196, j = p % 196, hp = j / 14, wp = j % 14;
    float s = pb[d];
    for (int c = 0; c < 3; ++c)
      for (int py = 0; py < 16; ++py) {
        const float* xr = x + ((((size_t)b * 3 + c) * 8 + t) * 224 + (hp * 16 + py)) * 224 + wp * 16;
        const float* wr = pw + (((size_t)d * 3 + c) * 16 + py) * 16;
        for (int qx = 0; qx < 16; ++qx) s += xr[qx] * wr[qx];
      }
    v = s + pos[(size_t)(1 + j) * DM + d] + tpos[(size_t)t * DM + d];
  }
  h[i] = v;
}

// ----------------------------------------------- residual add + RMSNorm -> f16
__global__ void k_add_rmsnorm(const float* __restrict__ hin, float* __restrict__ res,
                              const float* __restrict__ nw, _Float16* __restrict__ hn16,
                              int first) {
  int row = blockIdx.x, d = threadIdx.x;           // blockDim = 192
  size_t off = (size_t)row * DM + d;
  float v = 0.f;
  if (row < BL) {
    v = first ? hin[off] : (hin[off] + res[off]);
    res[off] = v;
  }
  __shared__ float red[DM];
  red[d] = v * v;
  __syncthreads();
  if (d < 64) red[d] = red[d] + red[d + 64] + red[d + 128];
  __syncthreads();
  for (int s = 32; s > 0; s >>= 1) { if (d < s) red[d] += red[d + s]; __syncthreads(); }
  float scale = rsqrtf(red[0] / (float)DM + 1e-5f);
  hn16[off] = (row < BL) ? (_Float16)(v * scale * nw[d]) : (_Float16)0.f;
}

// ------------------------------------------------------- WMMA f16 GEMM, C=A*B^T
// A: (M,K) f16 row-major, M multiple of 64; B: (N,K) f16 row-major; C: (M,N) f32.
// K multiple of 32, N multiple of 16. One wave computes 4 consecutive 16x16 M-tiles,
// reusing the B fragment across them (4 WMMA per B-fragment load).
// grid.z batches independent GEMMs via pointer strides.
__global__ void k_gemm_wmma(const _Float16* __restrict__ A, const _Float16* __restrict__ B,
                            float* __restrict__ C, int M, int N, int K,
                            size_t sA, size_t sB, size_t sC) {
  A += (size_t)blockIdx.z * sA;
  B += (size_t)blockIdx.z * sB;
  C += (size_t)blockIdx.z * sC;
  const int tn = blockIdx.y * blockDim.y + threadIdx.y;
  if (tn * 16 >= N) return;
  const int tm0 = blockIdx.x * 4;
  const int lane = threadIdx.x, grp = lane >> 4, idx = lane & 15;
  const _Float16* brow  = B + (size_t)(tn * 16 + idx) * K;
  const _Float16* arow0 = A + (size_t)(tm0 * 16 + idx) * K;
  v8f acc0 = {}, acc1 = {}, acc2 = {}, acc3 = {};
  for (int k0 = 0; k0 < K; k0 += 32) {
    v8h blo = *(const v8h*)(brow + k0 + grp * 8);
    v8h bhi = *(const v8h*)(brow + k0 + 16 + grp * 8);
    v16h b;
#pragma unroll
    for (int e = 0; e < 8; ++e) { b[e] = blo[e]; b[e + 8] = bhi[e]; }
#pragma unroll
    for (int mt = 0; mt < 4; ++mt) {
      const _Float16* arow = arow0 + (size_t)mt * 16 * K;
      v8h alo = *(const v8h*)(arow + k0 + grp * 8);
      v8h ahi = *(const v8h*)(arow + k0 + 16 + grp * 8);
      v16h a;
#pragma unroll
      for (int e = 0; e < 8; ++e) { a[e] = alo[e]; a[e + 8] = ahi[e]; }
      v8f& acc = (mt == 0) ? acc0 : (mt == 1) ? acc1 : (mt == 2) ? acc2 : acc3;
      acc = __builtin_amdgcn_wmma_f32_16x16x32_f16(false, a, false, b, (short)0, acc,
                                                   false, false);
    }
  }
#pragma unroll
  for (int mt = 0; mt < 4; ++mt) {
    const v8f acc = (mt == 0) ? acc0 : (mt == 1) ? acc1 : (mt == 2) ? acc2 : acc3;
#pragma unroll
    for (int v = 0; v < 8; ++v)
      C[(size_t)((tm0 + mt) * 16 + grp * 8 + v) * N + (tn * 16 + idx)] = acc[v];
  }
}

// ---------------------------------- causal depthwise conv(4) + SiLU, both dirs
__global__ void k_conv_silu(const float* __restrict__ xz,
                            const float* __restrict__ cwf, const float* __restrict__ cbf,
                            const float* __restrict__ cwb, const float* __restrict__ cbb,
                            float* __restrict__ xcf, float* __restrict__ xcb,
                            _Float16* __restrict__ xcf16, _Float16* __restrict__ xcb16) {
  int i = blockIdx.x * blockDim.x + threadIdx.x;
  if (i >= MPAD * DI) return;
  int c = i % DI, row = i / DI;
  if (row >= BL) { xcf16[i] = (_Float16)0.f; xcb16[i] = (_Float16)0.f; return; }
  int b = row / LSEQ, l = row % LSEQ;
  const float* xb = xz + (size_t)b * LSEQ * DXZ;
  float sf = cbf[c], sb = cbb[c];
#pragma unroll
  for (int k = 0; k < 4; ++k) {
    int lf = l + k - 3;
    if (lf >= 0) {
      sf += cwf[c * 4 + k] * xb[(size_t)lf * DXZ + c];                 // forward seq
      sb += cwb[c * 4 + k] * xb[(size_t)(LSEQ - 1 - lf) * DXZ + c];   // reversed seq
    }
  }
  float vf = sf / (1.f + __expf(-sf));
  float vb = sb / (1.f + __expf(-sb));
  xcf[i] = vf; xcb[i] = vb;
  xcf16[i] = (_Float16)vf; xcb16[i] = (_Float16)vb;
}

// ---------------------- dt = softplus(dbl[:12] @ W^T + b), both dirs in one grid
__global__ void k_dtproj(const float* __restrict__ dblbuf,
                         const float* __restrict__ wF, const float* __restrict__ wB,
                         const float* __restrict__ bF, const float* __restrict__ bB,
                         float* __restrict__ dtbuf) {
  int i = blockIdx.x * blockDim.x + threadIdx.x;
  if (i >= 2 * BL * DI) return;
  int d = i / (BL * DI);
  int r = i % (BL * DI);
  int c = r % DI, row = r / DI;
  const float* w    = d ? wB : wF;
  const float* bias = d ? bB : bF;
  const float* dr = dblbuf + (size_t)d * MPAD * NXP + (size_t)row * NXP;
  const float* wr = w + (size_t)c * DTR;
  float s = bias[c];
#pragma unroll
  for (int k = 0; k < DTR; ++k) s += dr[k] * wr[k];
  dtbuf[(size_t)d * MPAD * DI + (size_t)row * DI + c] = (s > 20.f) ? s : log1pf(__expf(s));
}

// -------------------------------------------------------- selective scan
// grid = 4 blocks: blockIdx.x = dir*2 + batch; block = DI threads.
// 16 states in registers/channel; per-step B/C (32 floats) fetched by each wave
// and distributed via lane shuffles -- no inter-wave barriers on the serial path.
__global__ void k_scan(const float* __restrict__ dtbuf, const float* __restrict__ xcbuf,
                       const float* __restrict__ dblbuf,
                       const float* __restrict__ AlogF, const float* __restrict__ AlogB,
                       const float* __restrict__ DF, const float* __restrict__ DB,
                       const float* __restrict__ xz, float* __restrict__ ybuf) {
  const int d = blockIdx.x >> 1, b = blockIdx.x & 1;
  const int c = threadIdx.x, lane = c & 31;
  const float* dtb  = dtbuf  + (size_t)d * MPAD * DI;
  const float* xc   = xcbuf  + (size_t)d * MPAD * DI;
  const float* dbl  = dblbuf + (size_t)d * MPAD * NXP;
  float*       y    = ybuf   + (size_t)d * MPAD * DI;
  const float* Alog = d ? AlogB : AlogF;
  const float* Dp   = d ? DB : DF;
  float A[DST], hst[DST];
#pragma unroll
  for (int s = 0; s < DST; ++s) { A[s] = -__expf(Alog[(size_t)c * DST + s]); hst[s] = 0.f; }
  float Dv = Dp[c];
  for (int l = 0; l < LSEQ; ++l) {
    size_t row = (size_t)b * LSEQ + l;
    float bc = dbl[row * NXP + DTR + lane];   // lanes 0-15: B, lanes 16-31: C
    float dt = dtb[row * DI + c];
    float u  = xc[row * DI + c];
    float du = dt * u;
    float acc = 0.f;
#pragma unroll
    for (int s = 0; s < DST; ++s) {
      float Bs = __shfl(bc, s, 32);
      float Cs = __shfl(bc, s + 16, 32);
      hst[s] = __expf(dt * A[s]) * hst[s] + du * Bs;
      acc += hst[s] * Cs;
    }
    acc += u * Dv;
    int zl = d ? (LSEQ - 1 - l) : l;
    float z = xz[((size_t)b * LSEQ + zl) * DXZ + DI + c];
    acc *= z / (1.f + __expf(-z));
    y[row * DI + c] = acc;
  }
}

// --------------------------------------- combine y_f + reverse(y_b) -> f16
__global__ void k_combine(const float* __restrict__ yf, const float* __restrict__ yb,
                          _Float16* __restrict__ yc16) {
  int i = blockIdx.x * blockDim.x + threadIdx.x;
  if (i >= MPAD * DI) return;
  int c = i % DI, row = i / DI;
  if (row >= BL) { yc16[i] = (_Float16)0.f; return; }
  int b = row / LSEQ, l = row % LSEQ;
  float v = yf[i] + yb[((size_t)b * LSEQ + (LSEQ - 1 - l)) * DI + c];
  yc16[i] = (_Float16)v;
}

// --------------------------------------------- final rmsnorm(token0) + head
__global__ void k_final_head(const float* __restrict__ h, const float* __restrict__ res,
                             const float* __restrict__ nfw, const float* __restrict__ hw,
                             const float* __restrict__ hb, float* __restrict__ out) {
  int b = blockIdx.x, t = threadIdx.x;     // blockDim = 256
  size_t row = (size_t)b * LSEQ;
  __shared__ float fin[DM];
  __shared__ float red[256];
  float v = 0.f;
  if (t < DM) v = h[row * DM + t] + res[row * DM + t];
  red[t] = v * v;
  __syncthreads();
  for (int s = 128; s > 0; s >>= 1) { if (t < s) red[t] += red[t + s]; __syncthreads(); }
  float scale = rsqrtf(red[0] / (float)DM + 1e-5f);
  if (t < DM) fin[t] = v * scale * nfw[t];
  __syncthreads();
  for (int cls = t; cls < 1000; cls += 256) {
    float s = hb[cls];
    const float* wr = hw + (size_t)cls * DM;
    for (int d = 0; d < DM; ++d) s += fin[d] * wr[d];
    out[b * 1000 + cls] = s;
  }
}

// ============================================================== host driver
static char* bump(char*& p, size_t bytes) {
  char* r = p;
  p += (bytes + 255) & ~(size_t)255;
  return r;
}

extern "C" void kernel_launch(void* const* d_in, const int* in_sizes, int n_in,
                              void* d_out, int out_size, void* d_ws, size_t ws_size,
                              hipStream_t stream) {
  (void)in_sizes; (void)n_in; (void)out_size;
  const float* x        = (const float*)d_in[0];
  const float* patch_w  = (const float*)d_in[1];
  const float* patch_b  = (const float*)d_in[2];
  const float* cls_tok  = (const float*)d_in[3];
  const float* pos_emb  = (const float*)d_in[4];
  const float* temp_pos = (const float*)d_in[5];
  const float* norm_w   = (const float*)d_in[6];
  const float* in_pw    = (const float*)d_in[7];
  const float* out_pw   = (const float*)d_in[8];
  const float* norm_fw  = (const float*)d_in[9];
  const float* head_w   = (const float*)d_in[10];
  const float* head_b   = (const float*)d_in[11];
  const float* conv_w[2]  = { (const float*)d_in[12], (const float*)d_in[19] };
  const float* conv_b[2]  = { (const float*)d_in[13], (const float*)d_in[20] };
  const float* xproj_w[2] = { (const float*)d_in[14], (const float*)d_in[21] };
  const float* dtp_w[2]   = { (const float*)d_in[15], (const float*)d_in[22] };
  const float* dtp_b[2]   = { (const float*)d_in[16], (const float*)d_in[23] };
  const float* A_log[2]   = { (const float*)d_in[17], (const float*)d_in[24] };
  const float* Dpar[2]    = { (const float*)d_in[18], (const float*)d_in[25] };

  char* p = (char*)d_ws;
  float*    res     = (float*)bump(p, (size_t)MPAD * DM * 4);
  float*    hbuf    = (float*)bump(p, (size_t)MPAD * DM * 4);
  float*    xz      = (float*)bump(p, (size_t)MPAD * DXZ * 4);
  float*    xcbuf   = (float*)bump(p, (size_t)2 * MPAD * DI * 4);    // dir-major
  float*    dtbuf   = (float*)bump(p, (size_t)2 * MPAD * DI * 4);
  float*    dblbuf  = (float*)bump(p, (size_t)2 * MPAD * NXP * 4);
  float*    ybuf    = (float*)bump(p, (size_t)2 * MPAD * DI * 4);
  _Float16* hn16    = (_Float16*)bump(p, (size_t)MPAD * DM * 2);
  _Float16* xc16buf = (_Float16*)bump(p, (size_t)2 * MPAD * DI * 2);
  _Float16* yc16    = (_Float16*)bump(p, (size_t)MPAD * DI * 2);
  _Float16* w_in16  = (_Float16*)bump(p, (size_t)DEPTH * DXZ * DM * 2);
  _Float16* w_out16 = (_Float16*)bump(p, (size_t)DEPTH * DM * DI * 2);
  _Float16* w_xp16  = (_Float16*)bump(p, (size_t)2 * DEPTH * NXP * DI * 2);  // dir-major
  if ((size_t)(p - (char*)d_ws) > ws_size) return;  // workspace too small

  // -------- weight conversion (deterministic, every call)
  {
    int n1 = DEPTH * DXZ * DM;
    k_cvt_f16<<<(n1 + 255) / 256, 256, 0, stream>>>(in_pw, w_in16, n1);
    int n2 = DEPTH * DM * DI;
    k_cvt_f16<<<(n2 + 255) / 256, 256, 0, stream>>>(out_pw, w_out16, n2);
    int n3 = DEPTH * NXP * DI;
    k_cvt_xproj<<<(n3 + 255) / 256, 256, 0, stream>>>(xproj_w[0], w_xp16);
    k_cvt_xproj<<<(n3 + 255) / 256, 256, 0, stream>>>(xproj_w[1], w_xp16 + (size_t)DEPTH * NXP * DI);
  }

  // -------- patch embed + sequence assembly
  k_patch_embed<<<(BL * DM + 255) / 256, 256, 0, stream>>>(
      x, patch_w, patch_b, cls_tok, pos_emb, temp_pos, hbuf);

  const dim3 blk2d(32, 4);
  const int  MT4 = MPAD / 64;  // 50 blocks of 4 M-tiles
  for (int i = 0; i < DEPTH; ++i) {
    k_add_rmsnorm<<<MPAD, DM, 0, stream>>>(hbuf, res, norm_w + (size_t)i * DM, hn16, i == 0);

    // xz = hn @ in_proj^T : (MPAD,192) x (768,192)^T
    k_gemm_wmma<<<dim3(MT4, DXZ / 64, 1), blk2d, 0, stream>>>(
        hn16, w_in16 + (size_t)i * DXZ * DM, xz, MPAD, DXZ, DM, 0, 0, 0);

    k_conv_silu<<<(MPAD * DI + 255) / 256, 256, 0, stream>>>(
        xz, conv_w[0] + (size_t)i * DI * 4, conv_b[0] + (size_t)i * DI,
        conv_w[1] + (size_t)i * DI * 4, conv_b[1] + (size_t)i * DI,
        xcbuf, xcbuf + (size_t)MPAD * DI, xc16buf, xc16buf + (size_t)MPAD * DI);

    // dbl = xc @ xproj^T for both dirs in one batched launch (grid.z = 2)
    k_gemm_wmma<<<dim3(MT4, 1, 2), blk2d, 0, stream>>>(
        xc16buf, w_xp16 + (size_t)i * NXP * DI, dblbuf, MPAD, NXP, DI,
        (size_t)MPAD * DI, (size_t)DEPTH * NXP * DI, (size_t)MPAD * NXP);

    k_dtproj<<<(2 * BL * DI + 255) / 256, 256, 0, stream>>>(
        dblbuf, dtp_w[0] + (size_t)i * DI * DTR, dtp_w[1] + (size_t)i * DI * DTR,
        dtp_b[0] + (size_t)i * DI, dtp_b[1] + (size_t)i * DI, dtbuf);

    k_scan<<<4, DI, 0, stream>>>(
        dtbuf, xcbuf, dblbuf,
        A_log[0] + (size_t)i * DI * DST, A_log[1] + (size_t)i * DI * DST,
        Dpar[0] + (size_t)i * DI, Dpar[1] + (size_t)i * DI, xz, ybuf);

    k_combine<<<(MPAD * DI + 255) / 256, 256, 0, stream>>>(
        ybuf, ybuf + (size_t)MPAD * DI, yc16);

    // h = ycomb @ out_proj^T : (MPAD,384) x (192,384)^T
    k_gemm_wmma<<<dim3(MT4, DM / 64, 1), blk2d, 0, stream>>>(
        yc16, w_out16 + (size_t)i * DM * DI, hbuf, MPAD, DM, DI, 0, 0, 0);
  }

  k_final_head<<<NB, 256, 0, stream>>>(hbuf, res, norm_fw, head_w, head_b, (float*)d_out);
}
